// Tacotron2_48103633715606
// MI455X (gfx1250) — matrix operand order repaired and checked by reading, not compiled, not run on hardware
//
#include <hip/hip_runtime.h>
#include <cstdint>
#include <cstddef>

// ---------------- model dims ----------------
constexpr int B_   = 64;
constexpr int T_   = 1024;
constexpr int H_   = 256;     // LSTM hidden
constexpr int DTXT = 512;
constexpr int DSPK = 64;
constexpr int DIN  = 576;     // 512 + 64
constexpr int NM   = 80;      // mels
constexpr int G4   = 1024;    // 4*H
constexpr int MROWS = T_ * B_;   // 65536

typedef __attribute__((ext_vector_type(16))) __bf16 bf16x16;
typedef __attribute__((ext_vector_type(8)))  __bf16 bf16x8;
typedef __attribute__((ext_vector_type(8)))  float  f32x8;

#define DEV __device__ __forceinline__

DEV f32x8 wmma_bf16(bf16x16 a, bf16x16 b, f32x8 c) {
  // D = A(16x32 bf16) * B(32x16 bf16) + C(16x16 f32)
  return __builtin_amdgcn_wmma_f32_16x16x32_bf16(false, a, false, b, (short)0, c,
                                                 false, false);
}

// A fragment: 16x32 tile of row-major [*, lda] bf16 matrix at (m0, k0).
// contiguous k order (same as load_b_frag): lanes 0-15 row=lane K = k0..k0+15;
//             lanes 16-31 row=lane-16 K = k0+16..k0+31.
DEV bf16x16 load_a_frag(const __bf16* A, int lda, int m0, int k0) {
  int l = threadIdx.x & 31;
  const __bf16* p = A + (size_t)(m0 + (l & 15)) * lda + (k0 + ((l >> 4) << 4));
  return *(const bf16x16*)p;
}

// B fragment: 32x16 (KxN) tile taken from row-major weight W[N][K] (i.e. B = W^T).
// lanes 0-15: column n = n0+lane, K = k0..k0+15 ; lanes 16-31: same n, K = k0+16..k0+31
DEV bf16x16 load_b_frag(const __bf16* W, int ldw, int n0, int k0) {
  int l = threadIdx.x & 31;
  const __bf16* p = W + (size_t)(n0 + (l & 15)) * ldw + (k0 + ((l >> 4) << 4));
  return *(const bf16x16*)p;
}

DEV float sigf(float x) { return 1.0f / (1.0f + __expf(-x)); }

// ---------------- small prep kernels ----------------
__global__ void k_cvt(const float* __restrict__ in, __bf16* __restrict__ out, int n) {
  int i = blockIdx.x * blockDim.x + threadIdx.x;
  if (i < n) out[i] = (__bf16)in[i];
}

__global__ void k_bias_add(const float* __restrict__ a, const float* __restrict__ b,
                           float* __restrict__ o, int n) {
  int i = blockIdx.x * blockDim.x + threadIdx.x;
  if (i < n) o[i] = a[i] + b[i];
}

// Split decoder Wih_d [1024][592] into Wenc [1024][512] and zero-padded Wmel [1024][96]
__global__ void k_wd_split(const float* __restrict__ Wd, __bf16* __restrict__ Wenc,
                           __bf16* __restrict__ Wmel) {
  int i = blockIdx.x * blockDim.x + threadIdx.x;
  const int CT = DTXT + 96;
  if (i >= G4 * CT) return;
  int row = i / CT, c = i % CT;
  if (c < DTXT) {
    Wenc[(size_t)row * DTXT + c] = (__bf16)Wd[(size_t)row * 592 + 80 + c];
  } else {
    int k = c - DTXT;
    Wmel[(size_t)row * 96 + k] = (k < NM) ? (__bf16)Wd[(size_t)row * 592 + k]
                                          : (__bf16)0.0f;
  }
}

// Build time-major encoder input X[T][B][576] in bf16
__global__ void k_embed(const int* __restrict__ seq, const int* __restrict__ spk,
                        const float* __restrict__ et, const float* __restrict__ es,
                        __bf16* __restrict__ X) {
  int i = blockIdx.x * blockDim.x + threadIdx.x;
  if (i >= T_ * B_ * DIN) return;
  int k = i % DIN;
  int tb = i / DIN;
  int b = tb % B_, t = tb / B_;
  float v;
  if (k < DTXT) {
    int tok = seq[b * T_ + t];
    v = et[(size_t)tok * DTXT + k];
  } else {
    v = es[(size_t)spk[b] * DSPK + (k - DTXT)];
  }
  X[i] = (__bf16)v;
}

// ---------------- big GEMM: C[M,1024] = A[M,K] * W[1024,K]^T + bias ----------------
// grid = (N/128, M/16), block = 256 (8 waves, one 16x16 C tile each)
__global__ __launch_bounds__(256) void k_gemm(const __bf16* __restrict__ A,
                                              const __bf16* __restrict__ W,
                                              const float* __restrict__ bias,
                                              __bf16* __restrict__ C, int K, int N) {
  int lane = threadIdx.x & 31, wave = threadIdx.x >> 5;
  int m0 = blockIdx.y << 4;
  int n0 = ((blockIdx.x << 3) + wave) << 4;
  f32x8 acc;
  float bv = bias[n0 + (lane & 15)];
#pragma unroll
  for (int r = 0; r < 8; ++r) acc[r] = bv;
  for (int k0 = 0; k0 < K; k0 += 32) {
    __builtin_prefetch((const void*)(A + (size_t)m0 * K + k0 + 128), 0, 1);
    bf16x16 a = load_a_frag(A, K, m0, k0);
    bf16x16 b = load_b_frag(W, K, n0, k0);
    acc = wmma_bf16(a, b, acc);
  }
  int n = n0 + (lane & 15), mb = m0 + ((lane >> 4) << 3);
#pragma unroll
  for (int r = 0; r < 8; ++r) C[(size_t)(mb + r) * N + n] = (__bf16)acc[r];
}

// ---------------- persistent bidirectional LSTM recurrence ----------------
// grid = 2 (dir), block = 1024 (32 waves). h double-buffered in LDS, c in VGPRs.
__global__ __launch_bounds__(1024) void k_lstm(const __bf16* __restrict__ Whh2,
                                               const __bf16* __restrict__ pre2,
                                               __bf16* __restrict__ enc) {
  int dir = blockIdx.x;
  const __bf16* W = Whh2 + (size_t)dir * G4 * H_;
  const __bf16* P = pre2 + (size_t)dir * (size_t)MROWS * G4;
  __shared__ __bf16 hbuf[2][B_][H_];   // 64 KB, double buffered
  int tid = threadIdx.x;
  for (int i = tid; i < 2 * B_ * H_; i += 1024) ((__bf16*)hbuf)[i] = (__bf16)0.0f;
  __syncthreads();

  int lane = tid & 31, wave = tid >> 5;
  int n_ = lane & 15, mbo = (lane >> 4) << 3;
  f32x8 cst[2];
#pragma unroll
  for (int p = 0; p < 2; ++p)
#pragma unroll
    for (int r = 0; r < 8; ++r) cst[p][r] = 0.0f;

  for (int s = 0; s < T_; ++s) {
    int t = dir ? (T_ - 1 - s) : s;
    const __bf16* hb = &hbuf[s & 1][0][0];
#pragma unroll
    for (int p = 0; p < 2; ++p) {
      int pid = wave * 2 + p;              // 0..63
      int m0 = (pid >> 4) << 4;            // batch tile
      int j0 = (pid & 15) << 4;            // hidden-unit tile
      int mb = m0 + mbo;
      f32x8 acc[4];
#pragma unroll
      for (int g = 0; g < 4; ++g) {
        int n = g * H_ + j0 + n_;
#pragma unroll
        for (int r = 0; r < 8; ++r)
          acc[g][r] = (float)P[((size_t)t * B_ + mb + r) * G4 + n];
      }
#pragma unroll
      for (int k0 = 0; k0 < H_; k0 += 32) {
        __builtin_prefetch((const void*)(W + (size_t)j0 * H_ + k0), 0, 1);
        bf16x16 a = load_a_frag(hb, H_, m0, k0);
#pragma unroll
        for (int g = 0; g < 4; ++g)
          acc[g] = wmma_bf16(a, load_b_frag(W, H_, g * H_ + j0, k0), acc[g]);
      }
      // fused LSTM pointwise: i,f,g,o -> c,h
#pragma unroll
      for (int r = 0; r < 8; ++r) {
        float iv = sigf(acc[0][r]);
        float fv = sigf(acc[1][r]);
        float gv = tanhf(acc[2][r]);
        float ov = sigf(acc[3][r]);
        float cn = fv * cst[p][r] + iv * gv;
        cst[p][r] = cn;
        float hv = ov * tanhf(cn);
        hbuf[(s + 1) & 1][mb + r][j0 + n_] = (__bf16)hv;
        enc[((size_t)t * B_ + mb + r) * (2 * H_) + dir * H_ + j0 + n_] = (__bf16)hv;
      }
    }
    __syncthreads();
  }
}

// ---------------- persistent decoder ----------------
// gates = E[s] + prev_mel @ Wmel^T ; c=sig(i)*tanh(g); h=sig(o)*tanh(c);
// out = h @ Wfc^T + bfc ; prev_mel = out.  grid=1, block=1024.
__global__ __launch_bounds__(1024) void k_decoder(const __bf16* __restrict__ E,
                                                  const __bf16* __restrict__ Wmel,
                                                  const __bf16* __restrict__ Wfc,
                                                  const float* __restrict__ bfc,
                                                  const float* __restrict__ mel0,
                                                  float* __restrict__ out) {
  __shared__ __bf16 hbuf[B_][H_];      // 32 KB
  __shared__ __bf16 melbuf[B_][96];    // 12 KB (K padded 80 -> 96)
  int tid = threadIdx.x, lane = tid & 31, wave = tid >> 5;
  for (int i = tid; i < B_ * 96; i += 1024) {
    int row = i / 96, c = i % 96;
    melbuf[row][c] = (c < NM) ? (__bf16)mel0[(size_t)row * T_ * NM + c] : (__bf16)0.0f;
  }
  __syncthreads();

  int n_ = lane & 15, mbo = (lane >> 4) << 3;
  for (int s = 0; s < T_ - 1; ++s) {
    // phase 1: gate GEMM (i,g,o only; f is dead in the reference) + pointwise
#pragma unroll
    for (int p = 0; p < 2; ++p) {
      int pid = wave * 2 + p;
      int m0 = (pid >> 4) << 4, j0 = (pid & 15) << 4;
      int mb = m0 + mbo;
      f32x8 acc[3];
#pragma unroll
      for (int g = 0; g < 3; ++g) {
        int gn = (g == 0) ? 0 : (g + 1) * H_;   // 0, 2H, 3H
        int n = gn + j0 + n_;
#pragma unroll
        for (int r = 0; r < 8; ++r)
          acc[g][r] = (float)E[((size_t)s * B_ + mb + r) * G4 + n];
      }
#pragma unroll
      for (int k0 = 0; k0 < 96; k0 += 32) {
        bf16x16 a = load_a_frag(&melbuf[0][0], 96, m0, k0);
#pragma unroll
        for (int g = 0; g < 3; ++g) {
          int gn = (g == 0) ? 0 : (g + 1) * H_;
          acc[g] = wmma_bf16(a, load_b_frag(Wmel, 96, gn + j0, k0), acc[g]);
        }
      }
#pragma unroll
      for (int r = 0; r < 8; ++r) {
        float cv = sigf(acc[0][r]) * tanhf(acc[1][r]);
        float hv = sigf(acc[2][r]) * tanhf(cv);
        hbuf[mb + r][j0 + n_] = (__bf16)hv;
      }
    }
    __syncthreads();
    // phase 2: FC (64x80 = 4x5 tiles) on waves 0..19; writes out + next prev_mel
    if (wave < 20) {
      int m0 = (wave / 5) << 4, n0 = (wave % 5) << 4;
      int mb = m0 + mbo, n = n0 + n_;
      f32x8 acc;
      float bv = bfc[n];
#pragma unroll
      for (int r = 0; r < 8; ++r) acc[r] = bv;
#pragma unroll
      for (int k0 = 0; k0 < H_; k0 += 32)
        acc = wmma_bf16(load_a_frag(&hbuf[0][0], H_, m0, k0),
                        load_b_frag(Wfc, H_, n0, k0), acc);
#pragma unroll
      for (int r = 0; r < 8; ++r) {
        float v = acc[r];
        out[((size_t)(mb + r) * (T_ - 1) + s) * NM + n] = v;   // [B,1023,80]
        melbuf[mb + r][n] = (__bf16)v;
      }
    }
    __syncthreads();
  }
}

// ---------------- host side ----------------
static inline size_t alignup(size_t x) { return (x + 255) & ~(size_t)255; }

extern "C" void kernel_launch(void* const* d_in, const int* in_sizes, int n_in,
                              void* d_out, int out_size, void* d_ws, size_t ws_size,
                              hipStream_t stream) {
  const int*   text_seq = (const int*)d_in[0];
  const float* mel      = (const float*)d_in[1];
  const int*   spk      = (const int*)d_in[2];
  const float* emb_t    = (const float*)d_in[3];
  const float* emb_s    = (const float*)d_in[4];
  const float* Wih_f    = (const float*)d_in[5];
  const float* Whh_f    = (const float*)d_in[6];
  const float* bih_f    = (const float*)d_in[7];
  const float* bhh_f    = (const float*)d_in[8];
  const float* Wih_b    = (const float*)d_in[9];
  const float* Whh_b    = (const float*)d_in[10];
  const float* bih_b    = (const float*)d_in[11];
  const float* bhh_b    = (const float*)d_in[12];
  const float* Wih_d    = (const float*)d_in[13];
  const float* bih_d    = (const float*)d_in[14];
  const float* bhh_d    = (const float*)d_in[15];
  const float* Wfc_f32  = (const float*)d_in[16];
  const float* bfc      = (const float*)d_in[17];

  char* ws = (char*)d_ws;
  size_t off = 0;
  __bf16* X     = (__bf16*)(ws + off); off = alignup(off + (size_t)MROWS * DIN * 2);
  __bf16* pre   = (__bf16*)(ws + off); off = alignup(off + (size_t)2 * MROWS * G4 * 2);
  __bf16* enc   = (__bf16*)(ws + off); off = alignup(off + (size_t)MROWS * (2 * H_) * 2);
  __bf16* Epre  = (__bf16*)(ws + off); off = alignup(off + (size_t)MROWS * G4 * 2);
  __bf16* Wihb  = (__bf16*)(ws + off); off = alignup(off + (size_t)2 * G4 * DIN * 2);
  __bf16* Whhb  = (__bf16*)(ws + off); off = alignup(off + (size_t)2 * G4 * H_ * 2);
  __bf16* Wdenc = (__bf16*)(ws + off); off = alignup(off + (size_t)G4 * DTXT * 2);
  __bf16* Wdmel = (__bf16*)(ws + off); off = alignup(off + (size_t)G4 * 96 * 2);
  __bf16* Wfcb  = (__bf16*)(ws + off); off = alignup(off + (size_t)NM * H_ * 2);
  float*  bias_f = (float*)(ws + off); off = alignup(off + G4 * 4);
  float*  bias_b = (float*)(ws + off); off = alignup(off + G4 * 4);
  float*  bias_d = (float*)(ws + off); off = alignup(off + G4 * 4);

  auto blks = [](int n, int t) { return (n + t - 1) / t; };

  // weight conversions to bf16
  k_cvt<<<blks(G4 * DIN, 256), 256, 0, stream>>>(Wih_f, Wihb, G4 * DIN);
  k_cvt<<<blks(G4 * DIN, 256), 256, 0, stream>>>(Wih_b, Wihb + (size_t)G4 * DIN, G4 * DIN);
  k_cvt<<<blks(G4 * H_, 256), 256, 0, stream>>>(Whh_f, Whhb, G4 * H_);
  k_cvt<<<blks(G4 * H_, 256), 256, 0, stream>>>(Whh_b, Whhb + (size_t)G4 * H_, G4 * H_);
  k_cvt<<<blks(NM * H_, 256), 256, 0, stream>>>(Wfc_f32, Wfcb, NM * H_);
  k_wd_split<<<blks(G4 * (DTXT + 96), 256), 256, 0, stream>>>(Wih_d, Wdenc, Wdmel);
  k_bias_add<<<blks(G4, 256), 256, 0, stream>>>(bih_f, bhh_f, bias_f, G4);
  k_bias_add<<<blks(G4, 256), 256, 0, stream>>>(bih_b, bhh_b, bias_b, G4);
  k_bias_add<<<blks(G4, 256), 256, 0, stream>>>(bih_d, bhh_d, bias_d, G4);

  // encoder input (time-major) in bf16
  k_embed<<<blks(T_ * B_ * DIN, 256), 256, 0, stream>>>(text_seq, spk, emb_t, emb_s, X);

  // pre-gates for both directions: [T*B, 1024] = X @ Wih^T + (bih+bhh)
  dim3 ggrid(G4 / 128, MROWS / 16);
  k_gemm<<<ggrid, 256, 0, stream>>>(X, Wihb, bias_f, pre, DIN, G4);
  k_gemm<<<ggrid, 256, 0, stream>>>(X, Wihb + (size_t)G4 * DIN, bias_b,
                                    pre + (size_t)MROWS * G4, DIN, G4);

  // sequential bidirectional LSTM (one persistent WGP per direction)
  k_lstm<<<2, 1024, 0, stream>>>(Whhb, pre, enc);

  // decoder pre-gates: E = enc @ Wdenc^T + (bih_d+bhh_d)
  k_gemm<<<ggrid, 256, 0, stream>>>(enc, Wdenc, bias_d, Epre, DTXT, G4);

  // sequential decoder
  k_decoder<<<1, 1024, 0, stream>>>(Epre, Wdmel, Wfcb, bfc, mel, (float*)d_out);

  (void)in_sizes; (void)n_in; (void)out_size; (void)ws_size;
}
